// DIN_33285996544407
// MI455X (gfx1250) — compile-verified, matrix operand order checked by reading, not run on hardware
//
#include <hip/hip_runtime.h>
#include <hip/hip_bf16.h>
#include <math.h>

// ---------------- problem constants ----------------
#define BB   8192
#define SS   200
#define FF   128
#define EE   64
#define AA   64
#define H1   128
#define H2   64
#define H3   32
#define EPS  1e-5f
#define CS   8           // s-chunk size (200 = 25 * 8)
#define NCHUNK (SS / CS) // 25

typedef __attribute__((ext_vector_type(16))) _Float16 v16h;
typedef __attribute__((ext_vector_type(8)))  _Float16 v8h;
typedef __attribute__((ext_vector_type(8)))  float    v8f;

// workspace layout (bytes)
#define WS_WIKT  0        // _Float16 [64][128]  B-matrix of kh GEMM, row = a
#define WS_WIQ   16384    // float   [128][64]   Wi @ Wq
#define WS_BIQ   49152    // float   [64]
#define WS_BKB   49408    // float   [64]

// dynamic LDS layout (bytes)
#define L_BUF0   0                     // float [CS][16][FF]  64 KB
#define L_BUF1   65536                 // float [CS][16][FF]  64 KB
#define L_QH     131072                // float [16][64]
#define L_CAND   135168                // float [16][64]
#define L_SCORE  139264                // float [CS][16]
#define L_WGT    139776                // float [16][CS]
#define L_AW     140288                // float [16][CS]
#define L_MISC   140800                // resc[16], m[16], d[16], len[16]
#define SMEM_BYTES 141056

// ---------------------------------------------------------------------------
// Prep: fold Wi through Wk / Wq.
// ---------------------------------------------------------------------------
__global__ void din_prep_kernel(const float* __restrict__ Wi,
                                const float* __restrict__ bi,
                                const float* __restrict__ Wq,
                                const float* __restrict__ Wk,
                                _Float16* __restrict__ wikT,
                                float* __restrict__ wiq,
                                float* __restrict__ biq,
                                float* __restrict__ bkb) {
  const int a = blockIdx.x;     // 0..63
  const int f = threadIdx.x;    // 0..127
  float sk = 0.f, sq = 0.f;
  #pragma unroll 4
  for (int e = 0; e < EE; ++e) {
    float w = Wi[f * EE + e];
    sk = fmaf(w, Wk[e * AA + a], sk);
    sq = fmaf(w, Wq[e * AA + a], sq);
  }
  wikT[a * FF + f] = (_Float16)sk;
  wiq[f * AA + a]  = sq;
  if (a == 0 && f < AA) {
    float bq = 0.f, bk = 0.f;
    for (int e = 0; e < EE; ++e) {
      bq = fmaf(bi[e], Wq[e * AA + f], bq);
      bk = fmaf(bi[e], Wk[e * AA + f], bk);
    }
    biq[f] = bq;
    bkb[f] = bk;
  }
}

// B-fragment (16-bit, 32x16 K-major) from a global f16 row pointer.
// lane&15 = N col; lanes 0-15 hold K rel {0..7,16..23}, lanes 16-31 {8..15,24..31}.
__device__ __forceinline__ v16h load_frag_g16(const _Float16* base, int krel) {
  union { v16h v; v8h h[2]; } u;
  u.h[0] = *(const v8h*)(base + krel);
  u.h[1] = *(const v8h*)(base + krel + 16);
  return u.v;
}

// A-fragment from an f32 LDS row pointer (convert to f16 at use).
__device__ __forceinline__ v16h load_frag_f32(const float* base) {
  float4 a = *(const float4*)(base);
  float4 b = *(const float4*)(base + 4);
  float4 c = *(const float4*)(base + 16);
  float4 d = *(const float4*)(base + 20);
  v16h v;
  v[0]  = (_Float16)a.x; v[1]  = (_Float16)a.y; v[2]  = (_Float16)a.z; v[3]  = (_Float16)a.w;
  v[4]  = (_Float16)b.x; v[5]  = (_Float16)b.y; v[6]  = (_Float16)b.z; v[7]  = (_Float16)b.w;
  v[8]  = (_Float16)c.x; v[9]  = (_Float16)c.y; v[10] = (_Float16)c.z; v[11] = (_Float16)c.w;
  v[12] = (_Float16)d.x; v[13] = (_Float16)d.y; v[14] = (_Float16)d.z; v[15] = (_Float16)d.w;
  return v;
}

// Issue the async DMA for one s-chunk: 32 x b128 per thread, global -> LDS.
__device__ __forceinline__ void async_stage(const float* __restrict__ histf,
                                            int b0, int c, unsigned ldsbase, int t) {
  #pragma unroll 8
  for (int i = 0; i < 32; ++i) {
    int q  = t + 128 * i;          // float4 index within chunk
    int e0 = q * 4;                // float index
    int sp = e0 >> 11;             // s' within chunk
    int rb = (e0 >> 7) & 15;       // b within block
    int f  = e0 & 127;
    int gboff = (((b0 + rb) * SS + (c * CS + sp)) * FF + f) * 4;  // byte offset
    unsigned loff = ldsbase + (unsigned)e0 * 4u;
    asm volatile("global_load_async_to_lds_b128 %0, %1, %2"
                 :: "v"(loff), "v"(gboff), "s"(histf)
                 : "memory");
  }
}

// ---------------------------------------------------------------------------
// Main fused kernel: one workgroup = 16 batch rows, single async-pipelined
// pass over S with online softmax + feature-space pooling.
// ---------------------------------------------------------------------------
__global__ void __launch_bounds__(128)
din_main_kernel(const float* __restrict__ cand,      // [B,F]
                const float* __restrict__ histf,     // [B,S,F]
                const int*   __restrict__ hlen,      // [B]
                const _Float16* __restrict__ wikT,   // [A][F] f16
                const float* __restrict__ wiq,       // [F][A]
                const float* __restrict__ biq,       // [A]
                const float* __restrict__ bkb,       // [A]
                const float* __restrict__ Wv,        // [A]
                const float* __restrict__ Wi,        // [F][E]
                const float* __restrict__ bi,        // [E]
                const float* __restrict__ W1, const float* __restrict__ b1,
                const float* __restrict__ g1, const float* __restrict__ be1,
                const float* __restrict__ m1, const float* __restrict__ v1,
                const float* __restrict__ W2, const float* __restrict__ b2,
                const float* __restrict__ g2, const float* __restrict__ be2,
                const float* __restrict__ m2, const float* __restrict__ v2,
                const float* __restrict__ W3, const float* __restrict__ b3,
                const float* __restrict__ g3, const float* __restrict__ be3,
                const float* __restrict__ m3, const float* __restrict__ v3,
                const float* __restrict__ Wo, const float* __restrict__ bo,
                float* __restrict__ out) {
  extern __shared__ __align__(16) unsigned char dynsm[];
  float* bufA   = (float*)(dynsm + L_BUF0);
  float* bufB   = (float*)(dynsm + L_BUF1);
  float* sQh    = (float*)(dynsm + L_QH);
  float* sCand  = (float*)(dynsm + L_CAND);
  float* sScore = (float*)(dynsm + L_SCORE);   // [s'][b]
  float* sWgt   = (float*)(dynsm + L_WGT);     // [b][s']  attn weights
  float* sAw    = (float*)(dynsm + L_AW);      // [b][s']  avg-mask weights
  float* sResc  = (float*)(dynsm + L_MISC);
  float* sM     = sResc + 16;
  float* sD     = sResc + 32;
  int*   sLen   = (int*)(sResc + 48);

  const int t    = threadIdx.x;
  const int lane = t & 31;
  const int wave = t >> 5;
  const int b0   = blockIdx.x * 16;
  const int krel = (lane < 16) ? 0 : 8;

  // ---- per-lane constants + persistent B fragments (Wik) from global ----
  float wvr[4], bkr[4];
  #pragma unroll
  for (int n = 0; n < 4; ++n) {
    wvr[n] = Wv[n * 16 + (lane & 15)];
    bkr[n] = bkb[n * 16 + (lane & 15)];
  }
  v16h bfr[16];                               // [n][k4]
  #pragma unroll
  for (int n = 0; n < 4; ++n) {
    const _Float16* wrow = wikT + (n * 16 + (lane & 15)) * FF;
    #pragma unroll
    for (int k4 = 0; k4 < 4; ++k4)
      bfr[n * 4 + k4] = load_frag_g16(wrow + 32 * k4, krel);
  }

  // ---- preamble: qh = cand @ Wiq + biq ; cand_emb = cand @ Wi + bi ----
  for (int o = t; o < 16 * AA; o += 128) {
    int b = o >> 6, a = o & 63;
    const float* crow = cand + (size_t)(b0 + b) * FF;
    float aq = biq[a], ae = bi[a];
    #pragma unroll 4
    for (int f = 0; f < FF; ++f) {
      float cv = crow[f];
      aq = fmaf(cv, wiq[f * AA + a], aq);
      ae = fmaf(cv, Wi[f * EE + a], ae);
    }
    sQh[o]   = aq;
    sCand[o] = ae;
  }
  if (t < 16) {
    sLen[t] = hlen[b0 + t];
    sM[t] = -1e30f;
    sD[t] = 0.f;
  }

  // pooled feature accumulators: thread t owns feature column f = t
  float pAtt[16], pAvg[16];
  #pragma unroll
  for (int b = 0; b < 16; ++b) { pAtt[b] = 0.f; pAvg[b] = 0.f; }

  // ---- async pipeline prologue: DMA chunk 0 and 1 ----
  const unsigned lb0 = (unsigned)(uintptr_t)bufA;
  const unsigned lb1 = (unsigned)(uintptr_t)bufB;
  async_stage(histf, b0, 0, lb0, t);
  async_stage(histf, b0, 1, lb1, t);

  // ================= stream over history, double buffered =================
  for (int c = 0; c < NCHUNK; ++c) {
    const float* bp = (c & 1) ? bufB : bufA;
    // wait for this chunk's DMA (own wave: in-order; <=32 keeps next chunk in flight)
    if (c + 1 < NCHUNK) asm volatile("s_wait_asynccnt 0x20" ::: "memory");
    else                asm volatile("s_wait_asynccnt 0x0"  ::: "memory");
    __syncthreads();

    // --- B: kh = histfeat @ Wik via WMMA, fused score reduction ---
    for (int ss = wave; ss < CS; ss += 4) {
      const float* hrow = bp + (ss * 16 + (lane & 15)) * FF;
      v16h afr[4];
      #pragma unroll
      for (int k4 = 0; k4 < 4; ++k4)
        afr[k4] = load_frag_f32(hrow + 32 * k4 + krel);

      float part[8];
      #pragma unroll
      for (int r = 0; r < 8; ++r) part[r] = 0.f;

      #pragma unroll
      for (int n = 0; n < 4; ++n) {
        v8f acc = {};
        #pragma unroll
        for (int k4 = 0; k4 < 4; ++k4) {
          acc = __builtin_amdgcn_wmma_f32_16x16x32_f16(
              false, afr[k4], false, bfr[n * 4 + k4], (short)0, acc, false, false);
        }
        float wv = wvr[n], bk = bkr[n];
        #pragma unroll
        for (int r = 0; r < 8; ++r) {
          int brow = r + ((lane >> 4) << 3);
          float h = sQh[brow * AA + n * 16 + (lane & 15)] + acc[r] + bk;
          part[r] += fmaxf(h, 0.f) * wv;
        }
      }
      #pragma unroll
      for (int r = 0; r < 8; ++r) {
        float v = part[r];
        v += __shfl_xor(v, 1, 32);
        v += __shfl_xor(v, 2, 32);
        v += __shfl_xor(v, 4, 32);
        v += __shfl_xor(v, 8, 32);
        if ((lane & 15) == 0)
          sScore[ss * 16 + r + ((lane >> 4) << 3)] = v;
      }
    }
    __syncthreads();

    // --- C: online-softmax chunk update (one thread per batch row) ---
    if (t < 16) {
      int b = t, L = sLen[b];
      float mo = sM[b], mc = -1e30f;
      float sc[CS];
      #pragma unroll
      for (int j = 0; j < CS; ++j) {
        bool valid = (c * CS + j) < L;
        sc[j] = valid ? sScore[j * 16 + b] : -1e30f;
        sAw[b * CS + j] = valid ? 1.f : 0.f;
        mc = fmaxf(mc, sc[j]);
      }
      float mn = fmaxf(mo, mc);
      float resc = __expf(mo - mn);
      float wsum = 0.f;
      #pragma unroll
      for (int j = 0; j < CS; ++j) {
        float w = __expf(sc[j] - mn);   // -1e30 -> 0
        sWgt[b * CS + j] = w;
        wsum += w;
      }
      sD[b] = sD[b] * resc + wsum;
      sM[b] = mn;
      sResc[b] = resc;
    }
    __syncthreads();

    // --- D: feature-space pooling (thread t = feature column) ---
    #pragma unroll
    for (int b = 0; b < 16; ++b) pAtt[b] *= sResc[b];
    for (int ss = 0; ss < CS; ++ss) {
      #pragma unroll
      for (int b = 0; b < 16; ++b) {
        float h = bp[(ss * 16 + b) * FF + t];
        pAtt[b] = fmaf(sWgt[b * CS + ss], h, pAtt[b]);
        pAvg[b] = fmaf(sAw[b * CS + ss], h, pAvg[b]);
      }
    }
    __syncthreads();   // everyone done reading bp

    // --- issue DMA for chunk c+2 into the buffer just freed ---
    if (c + 2 < NCHUNK)
      async_stage(histf, b0, c + 2, (c & 1) ? lb1 : lb0, t);
  }

  // ================= tail: embeddings + MLP =================
  float* sFeatI = bufA;                         // [16][128]
  float* sFeatA = bufA + 16 * FF;               // [16][128]
  float* sXcat  = bufA + 32 * FF;               // [16][192]
  #pragma unroll
  for (int b = 0; b < 16; ++b) {
    sFeatI[b * FF + t] = pAtt[b] / sD[b];
    sFeatA[b * FF + t] = pAvg[b] / (float)sLen[b];
  }
  __syncthreads();

  for (int o = t; o < 16 * EE; o += 128) {
    int b = o >> 6, e = o & 63;
    float ai = bi[e], aa = bi[e];
    #pragma unroll 4
    for (int f = 0; f < FF; ++f) {
      float w = Wi[f * EE + e];
      ai = fmaf(sFeatI[b * FF + f], w, ai);
      aa = fmaf(sFeatA[b * FF + f], w, aa);
    }
    sXcat[b * 192 + e]       = ai;
    sXcat[b * 192 + 64 + e]  = sCand[b * EE + e];
    sXcat[b * 192 + 128 + e] = aa;
  }
  __syncthreads();

  float* sX1 = bufB;                            // [16][128]
  for (int o = t; o < 16 * H1; o += 128) {
    int b = o >> 7, h = o & 127;
    float acc = b1[h];
    #pragma unroll 4
    for (int k = 0; k < 192; ++k) acc = fmaf(sXcat[b * 192 + k], W1[k * H1 + h], acc);
    acc = (acc - m1[h]) * rsqrtf(v1[h] + EPS) * g1[h] + be1[h];
    sX1[o] = fmaxf(acc, 0.f);
  }
  __syncthreads();

  float* sX2 = bufB + 16 * H1;                  // [16][64]
  for (int o = t; o < 16 * H2; o += 128) {
    int b = o >> 6, h = o & 63;
    float acc = b2[h];
    #pragma unroll 4
    for (int k = 0; k < H1; ++k) acc = fmaf(sX1[b * H1 + k], W2[k * H2 + h], acc);
    acc = (acc - m2[h]) * rsqrtf(v2[h] + EPS) * g2[h] + be2[h];
    sX2[o] = fmaxf(acc, 0.f);
  }
  __syncthreads();

  float* sX3 = bufB + 16 * H1 + 16 * H2;        // [16][32]
  for (int o = t; o < 16 * H3; o += 128) {
    int b = o >> 5, h = o & 31;
    float acc = b3[h];
    #pragma unroll 4
    for (int k = 0; k < H2; ++k) acc = fmaf(sX2[b * H2 + k], W3[k * H3 + h], acc);
    acc = (acc - m3[h]) * rsqrtf(v3[h] + EPS) * g3[h] + be3[h];
    sX3[o] = fmaxf(acc, 0.f);
  }
  __syncthreads();

  if (t < 16) {
    int b = t;
    float acc = bo[0];
    #pragma unroll
    for (int k = 0; k < H3; ++k) acc = fmaf(sX3[b * H3 + k], Wo[k], acc);
    out[b0 + b] = 1.f / (1.f + __expf(-acc));
  }
}

// ---------------------------------------------------------------------------
extern "C" void kernel_launch(void* const* d_in, const int* in_sizes, int n_in,
                              void* d_out, int out_size, void* d_ws, size_t ws_size,
                              hipStream_t stream) {
  const float* cand  = (const float*)d_in[0];
  const float* histf = (const float*)d_in[1];
  const int*   hlen  = (const int*)d_in[2];
  const float* Wi = (const float*)d_in[3];
  const float* bi = (const float*)d_in[4];
  const float* Wq = (const float*)d_in[5];
  const float* Wk = (const float*)d_in[6];
  const float* Wv = (const float*)d_in[7];
  const float* W1 = (const float*)d_in[8];
  const float* b1 = (const float*)d_in[9];
  const float* g1 = (const float*)d_in[10];
  const float* be1 = (const float*)d_in[11];
  const float* m1 = (const float*)d_in[12];
  const float* v1 = (const float*)d_in[13];
  const float* W2 = (const float*)d_in[14];
  const float* b2 = (const float*)d_in[15];
  const float* g2 = (const float*)d_in[16];
  const float* be2 = (const float*)d_in[17];
  const float* m2 = (const float*)d_in[18];
  const float* v2 = (const float*)d_in[19];
  const float* W3 = (const float*)d_in[20];
  const float* b3 = (const float*)d_in[21];
  const float* g3 = (const float*)d_in[22];
  const float* be3 = (const float*)d_in[23];
  const float* m3 = (const float*)d_in[24];
  const float* v3 = (const float*)d_in[25];
  const float* Wo = (const float*)d_in[26];
  const float* bo = (const float*)d_in[27];

  char* ws = (char*)d_ws;
  _Float16* wikT = (_Float16*)(ws + WS_WIKT);
  float*    wiq  = (float*)(ws + WS_WIQ);
  float*    biq  = (float*)(ws + WS_BIQ);
  float*    bkb  = (float*)(ws + WS_BKB);

  din_prep_kernel<<<64, 128, 0, stream>>>(Wi, bi, Wq, Wk, wikT, wiq, biq, bkb);

  din_main_kernel<<<BB / 16, 128, SMEM_BYTES, stream>>>(
      cand, histf, hlen, wikT, wiq, biq, bkb, Wv, Wi, bi,
      W1, b1, g1, be1, m1, v1,
      W2, b2, g2, be2, m2, v2,
      W3, b3, g3, be3, m3, v3,
      Wo, bo, (float*)d_out);
}
